// MaskedTopKAttnPool_40853728920173
// MI455X (gfx1250) — matrix operand order repaired and checked by reading, not compiled, not run on hardware
//
#include <hip/hip_runtime.h>
#include <hip/hip_bf16.h>
#include <math.h>

// ---------------------------------------------------------------------------
// MaskedTopKAttnPool for MI455X (gfx1250, wave32, WMMA + TDM)
// B=32, T=2048, D=1024, H=1024.  Dominant cost: 65536x1024x1024 bf16 GEMM.
// ---------------------------------------------------------------------------

#define B_      32
#define T_      2048
#define D_      1024
#define H_      1024
#define M_      (B_ * T_)      // 65536 rows
#define BLOCK_M 128            // rows per WG; A tile = 128*1024*2B = 256KB LDS
#define NCHUNK  64             // N columns per chunk (4 x 16-wide subtiles)
#define KSLICE  128            // K per B-slice: 64*128*2B = 16KB (double-buffered)
#define NSLICES (D_ / KSLICE)  // 8 slices per n-chunk
#define NCHUNKS (H_ / NCHUNK)  // 16 n-chunks
#define TOT_SLICES (NCHUNKS * NSLICES)  // 128
#define MAXK_PAD 1024

typedef __attribute__((ext_vector_type(16))) __bf16 v16bf;
typedef __attribute__((ext_vector_type(8)))  float  v8f;
typedef __attribute__((ext_vector_type(4)))  unsigned int v4u;
typedef __attribute__((ext_vector_type(8)))  int  v8i;
typedef __attribute__((ext_vector_type(4)))  int  v4i;

#if defined(__has_builtin)
#if __has_builtin(__builtin_amdgcn_tensor_load_to_lds) && \
    __has_builtin(__builtin_amdgcn_s_wait_tensorcnt)
#define USE_TDM 1
#endif
#endif
#ifndef USE_TDM
#define USE_TDM 0
#endif

union Frag32 {                 // 32-byte fragment: two uint4 halves <-> v16bf
  struct { uint4 lo; uint4 hi; } q;
  v16bf v;
};

__device__ __forceinline__ unsigned short f32_to_bf16(float f) {
  unsigned int u = __float_as_uint(f);
  u += 0x7FFFu + ((u >> 16) & 1u);   // round-to-nearest-even
  return (unsigned short)(u >> 16);
}

#if USE_TDM
// Issue a TDM 2D tile load: 64 rows x 128 bf16 (256B) from w1t (row stride
// 1024 elem) into LDS at lds_byte_off.  D# bitfields per CDNA5 ISA ch.8.
__device__ __forceinline__ void tdm_stage_b(const unsigned short* gsrc,
                                            unsigned lds_byte_off) {
  unsigned long long ga = (unsigned long long)(const void*)gsrc;
  v4u g0;
  g0[0] = 1u;                                   // count=1, user descriptor
  g0[1] = lds_byte_off;                         // lds_addr
  g0[2] = (unsigned)(ga & 0xFFFFFFFFu);         // global_addr[31:0]
  g0[3] = (unsigned)((ga >> 32) & 0x01FFFFFFu)  // global_addr[56:32]
          | 0x80000000u;                        // type = 2 ("image")
  v8i g1;
  g1[0] = 0x00010000;                           // data_size=1 (2B), mask=0
  g1[1] = (int)((D_ & 0xFFFF) << 16);           // tensor_dim0[15:0] = 1024
  g1[2] = (int)(((D_ >> 16) & 0xFFFF) | ((H_ & 0xFFFF) << 16)); // tensor_dim1
  g1[3] = (int)(((H_ >> 16) & 0xFFFF) | ((KSLICE & 0xFFFF) << 16)); // tile_dim0=128
  g1[4] = NCHUNK;                               // tile_dim1=64, tile_dim2=0
  g1[5] = D_;                                   // tensor_dim0_stride[31:0]
  g1[6] = 0;
  g1[7] = 0;
  v4i gz = {0, 0, 0, 0};
#if defined(__clang_major__) && (__clang_major__ >= 23)
  v8i gz8 = {0, 0, 0, 0, 0, 0, 0, 0};
  __builtin_amdgcn_tensor_load_to_lds(g0, g1, gz, gz, gz8, 0);
#else
  __builtin_amdgcn_tensor_load_to_lds(g0, g1, gz, gz, 0);
#endif
}
#endif

// ---------------------------------------------------------------------------
// Kernel 0: W1 [D][H] f32  ->  W1t [H][D] bf16  (tiled 32x32 transpose)
// ---------------------------------------------------------------------------
__global__ void w1_transpose_bf16(const float* __restrict__ w1,
                                  unsigned short* __restrict__ w1t) {
  __shared__ float tile[32][33];
  const int h0 = blockIdx.x * 32;
  const int d0 = blockIdx.y * 32;
  const int tx = threadIdx.x;        // 0..31
  const int ty = threadIdx.y;        // 0..7
#pragma unroll
  for (int i = 0; i < 32; i += 8)
    tile[ty + i][tx] = w1[(size_t)(d0 + ty + i) * H_ + (h0 + tx)];
  __syncthreads();
#pragma unroll
  for (int i = 0; i < 32; i += 8)
    w1t[(size_t)(h0 + ty + i) * D_ + (d0 + tx)] = f32_to_bf16(tile[tx][ty + i]);
}

// ---------------------------------------------------------------------------
// Kernel 1: fused scorer  scores[m] = w2 . tanh(x[m]*W1 + b1)
//   - A tile (128 x 1024 bf16) resident in LDS at full K (256KB)
//   - B slices double-buffered (2 x 16KB), filled by the Tensor Data Mover
//     (wave 0 issues tensor_load_to_lds; s_wait_tensorcnt + barrier syncs),
//     overlapping the DMA with the WMMA block of the current slice.
// ---------------------------------------------------------------------------
__global__ void __launch_bounds__(256)
score_gemm(const float* __restrict__ x,
           const unsigned short* __restrict__ w1t,
           const float* __restrict__ b1,
           const float* __restrict__ w2,
           float* __restrict__ scores) {
  extern __shared__ char smem[];
  unsigned short* lds_a  = (unsigned short*)smem;                      // [128][1024]
  unsigned short* lds_b0 = (unsigned short*)(smem + BLOCK_M * D_ * 2); // [64][128]
  unsigned short* lds_b1 = lds_b0 + NCHUNK * KSLICE;                   // [64][128]

  const int tid  = threadIdx.x;
  const int lane = tid & 31;
  const int wave = tid >> 5;     // 0..7
  const int half = lane >> 4;    // 0 | 1
  const int lr   = lane & 15;
  const int m0   = blockIdx.x * BLOCK_M;

  // ---- Stage A: 128 rows x 1024 f32 -> bf16 into LDS (read x exactly once) -
  {
    const float4* xg = (const float4*)(x + (size_t)m0 * D_);
    for (int idx = tid; idx < BLOCK_M * (D_ / 4); idx += 256) {
      float4 v = xg[idx];
      ushort4 o;
      o.x = f32_to_bf16(v.x); o.y = f32_to_bf16(v.y);
      o.z = f32_to_bf16(v.z); o.w = f32_to_bf16(v.w);
      ((ushort4*)lds_a)[idx] = o;
    }
  }

  // ---- Prime the pipeline: slice 0 (n0=0,k0=0) into buffer 0 ---------------
#if USE_TDM
  if (wave == 0) {
    tdm_stage_b(w1t, (unsigned)(unsigned long long)(void*)lds_b0);
    __builtin_amdgcn_s_wait_tensorcnt(0);
  }
#else
  for (int idx = tid; idx < NCHUNK * (KSLICE / 8); idx += 256) {
    const int n = idx >> 4, c = idx & 15;
    uint4 v = *(const uint4*)(w1t + (size_t)n * D_ + c * 8);
    *(uint4*)(lds_b0 + n * KSLICE + c * 8) = v;
  }
#endif
  __syncthreads();

  float partial[8];
#pragma unroll
  for (int r = 0; r < 8; ++r) partial[r] = 0.0f;

  int p = 0;   // which B buffer the compute reads this stage
  for (int nc = 0; nc < NCHUNKS; ++nc) {
    const int n0 = nc * NCHUNK;
    v8f acc[4];
#pragma unroll
    for (int j = 0; j < 4; ++j)
#pragma unroll
      for (int e = 0; e < 8; ++e) acc[j][e] = 0.0f;

    for (int ks = 0; ks < NSLICES; ++ks) {
      const int s = nc * NSLICES + ks;
      const bool havenext = (s + 1 < TOT_SLICES);
      const int n1 = ((s + 1) >> 3) * NCHUNK;   // next slice coords
      const int k1 = ((s + 1) & 7) * KSLICE;
      const unsigned short* curb = p ? lds_b1 : lds_b0;
      unsigned short*       nxtb = p ? lds_b0 : lds_b1;

      // -- kick off the next slice fill (overlaps with the WMMA block) ------
#if USE_TDM
      if (havenext && wave == 0)
        tdm_stage_b(w1t + (size_t)n1 * D_ + k1,
                    (unsigned)(unsigned long long)(void*)nxtb);
#else
      uint4 rr[4];
      if (havenext) {
        const unsigned short* src = w1t + (size_t)n1 * D_ + k1;
#pragma unroll
        for (int i = 0; i < 4; ++i) {
          const int idx = tid + i * 256;
          const int n = idx >> 4, c = idx & 15;
          rr[i] = *(const uint4*)(src + (size_t)n * D_ + c * 8);
        }
      }
#endif

      // -- WMMA block: 4 K-steps x 4 N-subtiles out of LDS buffer p ---------
#pragma unroll
      for (int kk = 0; kk < KSLICE / 32; ++kk) {
        // A fragment: row = m0 + 16*wave + lr
        //   half 0: K {kb..kb+7, kb+16..kb+23}; half 1: +8 (ISA A layout)
        Frag32 af;
        const unsigned short* ap =
            lds_a + (size_t)(wave * 16 + lr) * D_ + (ks * KSLICE) + kk * 32 + 8 * half;
        af.q.lo = *(const uint4*)(ap);
        af.q.hi = *(const uint4*)(ap + 16);
#pragma unroll
        for (int j = 0; j < 4; ++j) {
          // B fragment: col = n0 + 16*j + lr; half 0: K {0..7, 16..23}, half 1: +8
          Frag32 bfr;
          const unsigned short* bp =
              curb + (size_t)(j * 16 + lr) * KSLICE + kk * 32 + 8 * half;
          bfr.q.lo = *(const uint4*)(bp);
          bfr.q.hi = *(const uint4*)(bp + 16);
          acc[j] = __builtin_amdgcn_wmma_f32_16x16x32_bf16(
              false, af.v, false, bfr.v, (short)0, acc[j], false, false);
        }
      }

      // -- land the next slice, one barrier per stage -----------------------
#if USE_TDM
      if (havenext && wave == 0)
        __builtin_amdgcn_s_wait_tensorcnt(0);
#else
      if (havenext) {
#pragma unroll
        for (int i = 0; i < 4; ++i) {
          const int idx = tid + i * 256;
          const int n = idx >> 4, c = idx & 15;
          *(uint4*)(nxtb + n * KSLICE + c * 8) = rr[i];
        }
      }
#endif
      __syncthreads();
      p ^= 1;
    }

    // ---- fused epilogue: partial[r] += w2[col] * tanh(acc + b1[col]) ------
    // C layout: (vgpr r, lane) = row(m0+16w + r + 8*half), col(n0+16j+lr)
#pragma unroll
    for (int j = 0; j < 4; ++j) {
      const int col = n0 + j * 16 + lr;
      const float bb = b1[col];
      const float ww = w2[col];
#pragma unroll
      for (int r = 0; r < 8; ++r)
        partial[r] += ww * tanhf(acc[j][r] + bb);
    }
  }

  // ---- reduce the 16 column-lanes within each half, write 8 rows/lane ------
#pragma unroll
  for (int r = 0; r < 8; ++r) {
    float v = partial[r];
    v += __shfl_xor(v, 1, 32);
    v += __shfl_xor(v, 2, 32);
    v += __shfl_xor(v, 4, 32);
    v += __shfl_xor(v, 8, 32);
    partial[r] = v;
  }
  if (lr == 0) {
    const int rowbase = m0 + wave * 16 + 8 * half;
#pragma unroll
    for (int r = 0; r < 8; ++r) scores[rowbase + r] = partial[r];
  }
}

// ---------------------------------------------------------------------------
// Kernel 2: per-batch masked bitonic sort (desc score, asc index on ties) and
// top-k index emission. Tie rule reproduces jax.lax.top_k incl. -inf padding.
// ---------------------------------------------------------------------------
__global__ void __launch_bounds__(256)
topk_sort(const float* __restrict__ scores,
          const long long* __restrict__ lengths,
          int* __restrict__ topk, int* __restrict__ kcount) {
  __shared__ float sv[T_];
  __shared__ int   si[T_];
  const int b   = blockIdx.x;
  const int tid = threadIdx.x;
  const int len = (int)lengths[b];

  for (int t = tid; t < T_; t += 256) {
    float s = scores[b * T_ + t];
    if (t >= len) s = -INFINITY;
    sv[t] = s; si[t] = t;
  }
  __syncthreads();

  for (int k = 2; k <= T_; k <<= 1) {
    for (int j = k >> 1; j > 0; j >>= 1) {
      for (int i = tid; i < T_; i += 256) {
        const int ixj = i ^ j;
        if (ixj > i) {
          float a = sv[i], c = sv[ixj];
          int  ia = si[i], ic = si[ixj];
          const bool ord_ca = (c > a) || (c == a && ic < ia); // c precedes a
          const bool ord_ac = (a > c) || (a == c && ia < ic); // a precedes c
          const bool dirDesc = ((i & k) == 0);
          if (dirDesc ? ord_ca : ord_ac) {
            sv[i] = c; sv[ixj] = a; si[i] = ic; si[ixj] = ia;
          }
        }
      }
      __syncthreads();
    }
  }

  int kv = (int)ceilf((float)len * 0.35f);
  kv = kv < 6 ? 6 : kv;
  kv = kv > T_ ? T_ : kv;
  for (int t = tid; t < kv; t += 256) topk[b * MAXK_PAD + t] = si[t];
  if (tid == 0) kcount[b] = kv;
}

// ---------------------------------------------------------------------------
// Kernel 3: pooled[b] = mean over top-k rows of x[b] (exact fp32)
// ---------------------------------------------------------------------------
__global__ void __launch_bounds__(256)
pool_gather(const float* __restrict__ x, const int* __restrict__ topk,
            const int* __restrict__ kcount, float* __restrict__ out) {
  const int b   = blockIdx.x;
  const int tid = threadIdx.x;
  const int kv  = kcount[b];
  const int d0  = tid * 4;
  float4 acc = make_float4(0.f, 0.f, 0.f, 0.f);
  for (int j = 0; j < kv; ++j) {
    const int row = topk[b * MAXK_PAD + j];
    const float4 v = *(const float4*)(x + ((size_t)b * T_ + row) * D_ + d0);
    acc.x += v.x; acc.y += v.y; acc.z += v.z; acc.w += v.w;
  }
  const float inv = 1.0f / (float)kv;
  float4 o = make_float4(acc.x * inv, acc.y * inv, acc.z * inv, acc.w * inv);
  *(float4*)(out + (size_t)b * D_ + d0) = o;
}

// ---------------------------------------------------------------------------
extern "C" void kernel_launch(void* const* d_in, const int* in_sizes, int n_in,
                              void* d_out, int out_size, void* d_ws, size_t ws_size,
                              hipStream_t stream) {
  const float*     x       = (const float*)d_in[0];
  const long long* lengths = (const long long*)d_in[1];   // jnp.int64
  const float*     w1      = (const float*)d_in[2];
  const float*     b1      = (const float*)d_in[3];
  const float*     w2      = (const float*)d_in[4];
  float*           out     = (float*)d_out;

  char* ws = (char*)d_ws;
  unsigned short* w1t    = (unsigned short*)ws;                              // 2 MB
  float*          scores = (float*)(ws + (size_t)2 * 1024 * 1024);           // 256 KB
  int*            topk   = (int*)(ws + (size_t)2 * 1024 * 1024 + 262144);    // 128 KB
  int*            kcount = (int*)(ws + (size_t)2 * 1024 * 1024 + 262144 + 131072);

  // 0) W1 -> bf16, transposed so B-fragments are K-contiguous
  w1_transpose_bf16<<<dim3(H_ / 32, D_ / 32), dim3(32, 8), 0, stream>>>(w1, w1t);

  // 1) fused bf16-WMMA scorer; 256KB A tile + 2x16KB TDM-fed B slices in LDS
  const int smem_bytes = BLOCK_M * D_ * 2 + 2 * NCHUNK * KSLICE * 2;  // 294912 B
  (void)hipFuncSetAttribute((const void*)score_gemm,
                            hipFuncAttributeMaxDynamicSharedMemorySize, smem_bytes);
  score_gemm<<<M_ / BLOCK_M, 256, smem_bytes, stream>>>(x, w1t, b1, w2, scores);

  // 2) per-batch masked top-k (bitonic in LDS)
  topk_sort<<<B_, 256, 0, stream>>>(scores, lengths, topk, kcount);

  // 3) gather + mean
  pool_gather<<<B_, 256, 0, stream>>>(x, topk, kcount, out);
}
